// DETRLoss_41429254537588
// MI455X (gfx1250) — compile-verified, matrix operand order checked by reading, not run on hardware
//
#include <hip/hip_runtime.h>
#include <hip/hip_bf16.h>

typedef __attribute__((ext_vector_type(16))) _Float16 v16h;
typedef __attribute__((ext_vector_type(8)))  float    v8f;

#define NIMG 32
#define QN   300
#define CN   92
#define TN   32
#define NGRP 19            // ceil(300/16) query groups of 16
#define BIG  1.0e30f

// ---------------------------------------------------------------------------
// Kernel 0: zero the scalar output (harness poisons d_out with 0xAA).
// ---------------------------------------------------------------------------
__global__ void detr_zero_out(float* out) {
    if (threadIdx.x == 0 && blockIdx.x == 0) out[0] = 0.0f;
}

// ---------------------------------------------------------------------------
// Kernel 1: per-image diagonal cost blocks.
// One wave (32 threads) per (image, 16-query group).
// Softmax denominators on the matrix pipe:
//   D = A(exp(l - max), f16, 16x32) x B(ones, 32x16), accumulated over 3 K-chunks.
// B is all ones, so any bijective packing of the 16x32 chunk into the A
// fragment yields correct row sums (D[m][n] = sum_k A[m][k]).
// Lane pair (m, m+16) serves query q0+m; each lane loads its 48 classes as
// 12 x b128 (rows are 16B aligned: 92 floats = 368 B), with the single OOB
// chunk (classes 92..95) clamped+masked branch-free to -BIG so exp() -> 0.
// Output layout: cost[img][t][q]  (32 x 32 x 300 floats) in d_ws.
// ---------------------------------------------------------------------------
__global__ __launch_bounds__(32) void detr_cost_kernel(
    const float* __restrict__ logits,   // [32][300][92]
    const float* __restrict__ pboxes,   // [32][300][4] cxcywh
    const int*   __restrict__ labels,   // [1024]
    const float* __restrict__ tboxes,   // [1024][4] cxcywh
    float*       __restrict__ cost)     // [32][32][300]
{
    const int img = blockIdx.x / NGRP;
    const int grp = blockIdx.x % NGRP;
    const int q0  = grp * 16;
    const int lid = threadIdx.x;
    const int m    = lid & 15;          // query row this lane pair serves
    const int half = lid >> 4;          // which half of the K range
    const int q    = q0 + m;
    const int qc   = (q < QN) ? q : (QN - 1);     // clamp pad queries (unused results)
    const float* lrow = logits + (size_t)(img * QN + qc) * CN;

    // ---- bulk load this lane's 48 classes: 12 x float4, no per-element waits ----
    float4 vv[12];
    #pragma unroll
    for (int i = 0; i < 12; ++i) {
        const int kc  = i >> 2, vec = i & 3;
        const int off = kc * 32 + half * 16 + vec * 4;
        const int offc = (off > CN - 4) ? (CN - 4) : off;   // clamp the 92..95 chunk to 88
        vv[i] = *(const float4*)(lrow + offc);
    }
    // branch-free tail mask: classes >= 92 become -BIG (exp underflows to 0)
    #pragma unroll
    for (int i = 0; i < 12; ++i) {
        const int kc  = i >> 2, vec = i & 3;
        const int off = kc * 32 + half * 16 + vec * 4;
        float4 w = vv[i];
        w.x = (off + 0 < CN) ? w.x : -BIG;
        w.y = (off + 1 < CN) ? w.y : -BIG;
        w.z = (off + 2 < CN) ? w.z : -BIG;
        w.w = (off + 3 < CN) ? w.w : -BIG;
        vv[i] = w;
    }

    // ---- row max from the registers (masked -BIG elements are neutral) ----
    float mx = -BIG;
    #pragma unroll
    for (int i = 0; i < 12; ++i) {
        const float4 w = vv[i];
        mx = fmaxf(mx, fmaxf(fmaxf(w.x, w.y), fmaxf(w.z, w.w)));
    }
    mx = fmaxf(mx, __shfl_xor(mx, 16));           // combine the two row halves

    // ---- exp-sum via V_WMMA_F32_16X16X32_F16 against a ones matrix ----
    v16h ones;
    #pragma unroll
    for (int k = 0; k < 16; ++k) ones[k] = (_Float16)1.0f;
    v8f acc = {};
    #pragma unroll
    for (int kc = 0; kc < 3; ++kc) {
        v16h a;
        #pragma unroll
        for (int vec = 0; vec < 4; ++vec) {
            const float4 w = vv[kc * 4 + vec];
            a[vec * 4 + 0] = (_Float16)__expf(w.x - mx);
            a[vec * 4 + 1] = (_Float16)__expf(w.y - mx);
            a[vec * 4 + 2] = (_Float16)__expf(w.z - mx);
            a[vec * 4 + 3] = (_Float16)__expf(w.w - mx);
        }
        acc = __builtin_amdgcn_wmma_f32_16x16x32_f16(
            /*neg_a=*/false, a, /*neg_b=*/false, ones,
            /*c_mod=*/(short)0, acc, /*reuse_a=*/false, /*reuse_b=*/false);
    }

    // C/D layout: vgpr r, lanes 0-15 -> M=r ; lanes 16-31 -> M=r+8 (N replicated).
    __shared__ float s_sum[16];
    __shared__ float s_max[16];
    if (lid == 0) {
        #pragma unroll
        for (int r = 0; r < 8; ++r) s_sum[r] = acc[r];
    }
    if (lid == 16) {
        #pragma unroll
        for (int r = 0; r < 8; ++r) s_sum[8 + r] = acc[r];
    }
    if (half == 0) s_max[m] = mx;
    __syncthreads();

    // ---- cost entries: lane = target t, loop over the 16 queries ----
    const int t   = lid;
    const int lab = labels[img * TN + t];
    const float4 tb = *(const float4*)(tboxes + (size_t)(img * TN + t) * 4);
    const float tcx = tb.x, tcy = tb.y, tw = tb.z, th = tb.w;
    const float tx0 = tcx - 0.5f * tw, ty0 = tcy - 0.5f * th;
    const float tx1 = tcx + 0.5f * tw, ty1 = tcy + 0.5f * th;
    const float areaB = tw * th;

    for (int mm = 0; mm < 16; ++mm) {
        const int qq = q0 + mm;
        if (qq >= QN) break;                       // uniform across the wave
        const float4 pb = *(const float4*)(pboxes + (size_t)(img * QN + qq) * 4);
        const float pcx = pb.x, pcy = pb.y, pw = pb.z, ph = pb.w;
        // L1 in cxcywh space
        const float l1 = fabsf(pcx - tcx) + fabsf(pcy - tcy)
                       + fabsf(pw  - tw ) + fabsf(ph  - th );
        // GIoU in xyxy space
        const float px0 = pcx - 0.5f * pw, py0 = pcy - 0.5f * ph;
        const float px1 = pcx + 0.5f * pw, py1 = pcy + 0.5f * ph;
        const float areaA = pw * ph;
        const float iw = fmaxf(fminf(px1, tx1) - fmaxf(px0, tx0), 0.0f);
        const float ih = fmaxf(fminf(py1, ty1) - fmaxf(py0, ty0), 0.0f);
        const float inter = iw * ih;
        const float uni   = areaA + areaB - inter;
        const float iou   = inter / uni;
        const float cw = fmaxf(fmaxf(px1, tx1) - fminf(px0, tx0), 0.0f);
        const float ch = fmaxf(fmaxf(py1, ty1) - fminf(py0, ty0), 0.0f);
        const float areaC = cw * ch;
        const float giou  = iou - (areaC - uni) / areaC;
        // class cost: -softmax prob of this target's label
        const float lg   = logits[(size_t)(img * QN + qq) * CN + lab];
        const float prob = __expf(lg - s_max[mm]) / s_sum[mm];
        const float c = 5.0f * l1 - prob - 2.0f * giou;
        cost[((size_t)(img * TN + t)) * QN + qq] = c;
    }
}

// ---------------------------------------------------------------------------
// Kernel 2: wave-parallel Jonker-Volgenant LSA per image + matched-cost sum.
// One wave per image; lane L owns columns j = L+1, L+33, ... (1..300);
// lane 0 additionally owns the virtual column 0.
// ---------------------------------------------------------------------------
__global__ __launch_bounds__(32) void detr_hungarian_kernel(
    const float* __restrict__ cost,     // [32][32][300]
    float*       __restrict__ out)
{
    const int img  = blockIdx.x;
    const int lane = threadIdx.x;
    const float* C = cost + (size_t)img * TN * QN;   // rows = targets (32), cols = queries (300)
    const int n = TN, mcols = QN;

    __shared__ float v[QN + 1], minv[QN + 1], u[TN + 1];
    __shared__ int   p[QN + 1], way[QN + 1], used[QN + 1];

    for (int j = lane; j <= mcols; j += 32) { v[j] = 0.0f; p[j] = 0; way[j] = 0; }
    for (int r = lane; r <= n; r += 32) u[r] = 0.0f;
    __syncthreads();

    for (int ii = 1; ii <= n; ++ii) {
        if (lane == 0) p[0] = ii;
        for (int j = lane + 1; j <= mcols; j += 32) { minv[j] = BIG; used[j] = 0; }
        if (lane == 0) used[0] = 0;
        __syncthreads();

        int j0 = 0;
        while (true) {
            if (lane == 0) used[j0] = 1;
            __syncthreads();
            const int   i0   = p[j0];
            const float ui0  = u[i0];
            const float* crow = C + (size_t)(i0 - 1) * QN;
            __builtin_prefetch(crow + lane * 8, 0, 0);   // global_prefetch_b8

            // update minv/way over this lane's free columns; local argmin
            float best  = BIG;
            int   bestj = 1 << 29;
            for (int j = lane + 1; j <= mcols; j += 32) {
                if (!used[j]) {
                    const float cur = crow[j - 1] - ui0 - v[j];
                    if (cur < minv[j]) { minv[j] = cur; way[j] = j0; }
                    if (minv[j] < best) { best = minv[j]; bestj = j; }
                }
            }
            // wave argmin, tie-break to smallest j (matches np.argmin)
            #pragma unroll
            for (int off = 16; off > 0; off >>= 1) {
                const float ob = __shfl_xor(best, off);
                const int   oj = __shfl_xor(bestj, off);
                if (ob < best || (ob == best && oj < bestj)) { best = ob; bestj = oj; }
            }
            const int   j1    = bestj;
            const float delta = best;

            // dual updates: distinct u indices across lanes (distinct p[j])
            for (int j = lane + 1; j <= mcols; j += 32) {
                if (used[j]) { u[p[j]] += delta; v[j] -= delta; }
                else         { minv[j] -= delta; }
            }
            if (lane == 0) { u[p[0]] += delta; v[0] -= delta; }
            __syncthreads();

            j0 = j1;
            if (p[j0] == 0) break;
        }

        // augment along the alternating path (sequential, lane 0)
        if (lane == 0) {
            while (j0) { const int j1 = way[j0]; p[j0] = p[j1]; j0 = j1; }
        }
        __syncthreads();
    }

    // sum of matched costs: column j (1..300) matched to row p[j]-1
    float s = 0.0f;
    for (int j = lane + 1; j <= mcols; j += 32)
        if (p[j] != 0) s += C[(size_t)(p[j] - 1) * QN + (j - 1)];
    #pragma unroll
    for (int off = 16; off > 0; off >>= 1) s += __shfl_xor(s, off);
    if (lane == 0) atomicAdd(out, s);
}

// ---------------------------------------------------------------------------
extern "C" void kernel_launch(void* const* d_in, const int* in_sizes, int n_in,
                              void* d_out, int out_size, void* d_ws, size_t ws_size,
                              hipStream_t stream) {
    const float* logits = (const float*)d_in[0];   // (32,300,92) f32
    const float* pboxes = (const float*)d_in[1];   // (32,300,4)  f32
    const int*   labels = (const int*)  d_in[2];   // (1024,)     i32
    const float* tboxes = (const float*)d_in[3];   // (1024,4)    f32
    float* out  = (float*)d_out;                   // scalar
    float* cost = (float*)d_ws;                    // 32*32*300 floats = 1.2 MB

    detr_zero_out<<<1, 1, 0, stream>>>(out);
    detr_cost_kernel<<<NIMG * NGRP, 32, 0, stream>>>(logits, pboxes, labels, tboxes, cost);
    detr_hungarian_kernel<<<NIMG, 32, 0, stream>>>(cost, out);
}
